// CommBlock_22703197126777
// MI455X (gfx1250) — matrix operand order repaired and checked by reading, not compiled, not run on hardware
//
#include <hip/hip_runtime.h>
#include <hip/hip_bf16.h>

typedef __attribute__((ext_vector_type(16))) _Float16 v16h;
typedef __attribute__((ext_vector_type(8)))  float    v8f;
typedef __attribute__((ext_vector_type(4)))  unsigned v4u;
typedef __attribute__((ext_vector_type(8)))  int      v8i;
typedef __attribute__((ext_vector_type(4)))  int      v4i;

#define WMMA_F16(a, b, c) \
  __builtin_amdgcn_wmma_f32_16x16x32_f16(false, (a), false, (b), (short)0, (c), false, false)

// ---------------------------------------------------------------------------
// Fragment loaders, contiguous k order: element s of lane half h holds k = 16h + s (wave32)
// ---------------------------------------------------------------------------
__device__ __forceinline__ int lane_id() { return (int)(threadIdx.x & 31u); }

// A fragment: 16x32 f16 tile (row-major, leading dim lda).
__device__ __forceinline__ v16h load_a_frag(const _Float16* __restrict__ A, int lda) {
  int lane = lane_id();
  int m = lane & 15, h = lane >> 4;
  const _Float16* row = A + (size_t)m * (size_t)lda;
  v16h r;
#pragma unroll
  for (int v = 0; v < 8; ++v) {
    int k = (h << 4) + 2 * v;
    r[2 * v]     = row[k];
    r[2 * v + 1] = row[k + 1];
  }
  return r;
}

// B fragment (B = W^T) from W stored row-major [n][k] (16 rows x 32 k).
__device__ __forceinline__ v16h load_b_frag(const _Float16* __restrict__ W, int ldw) {
  int lane = lane_id();
  int n = lane & 15, h = lane >> 4;
  const _Float16* row = W + (size_t)n * (size_t)ldw + (h << 4);
  v16h r;
#pragma unroll
  for (int v = 0; v < 8; ++v) {
    r[2 * v]     = row[2 * v];
    r[2 * v + 1] = row[2 * v + 1];
  }
  return r;
}

// B fragment from matrix stored [k][n] row-major (32 x 16 tile, leading dim ldn).
__device__ __forceinline__ v16h load_b_frag_kn(const _Float16* __restrict__ Bm, int ldn) {
  int lane = lane_id();
  int n = lane & 15, h = lane >> 4;
  v16h r;
#pragma unroll
  for (int v = 0; v < 8; ++v) {
    int k = (h << 4) + 2 * v;
    r[2 * v]     = Bm[(size_t)k * ldn + n];
    r[2 * v + 1] = Bm[(size_t)(k + 1) * ldn + n];
  }
  return r;
}

// C/D fragment store (f16 output)
__device__ __forceinline__ void store_c_f16(_Float16* __restrict__ C, int ldc, v8f c) {
  int lane = lane_id();
  int n = lane & 15, h = lane >> 4;
#pragma unroll
  for (int i = 0; i < 8; ++i)
    C[(size_t)(i + 8 * h) * ldc + n] = (_Float16)c[i];
}

// ---------------------------------------------------------------------------
// Tensor Data Mover: async 2D tile load (global -> LDS), f16 elements.
// D# per CDNA5 ISA; 6-arg builtin form
// (uint32x4 g0, int32x8 g1, int32x4 g2, int32x4 g3, int32x8 g4, i32 cpol).
// ---------------------------------------------------------------------------
#if __has_builtin(__builtin_amdgcn_tensor_load_to_lds)
#define HAVE_TDM 1
__device__ __forceinline__ void tdm_load_2d_f16(unsigned lds_off, const void* gptr,
                                                unsigned tile_d0, unsigned tile_d1,
                                                unsigned stride_d0_elems) {
  unsigned long long ga = (unsigned long long)(size_t)gptr;
  v4u g0;
  g0[0] = 1u;                                        // count=1, user descriptor
  g0[1] = lds_off;                                   // LDS byte address
  g0[2] = (unsigned)(ga & 0xFFFFFFFFu);              // global_addr[31:0]
  g0[3] = (unsigned)((ga >> 32) & 0x01FFFFFFu) | (2u << 30); // addr[56:32], type=2
  v8i g1;
  g1[0] = (int)(1u << 16);                           // data_size=1 (2 bytes)
  g1[1] = (int)((tile_d0 & 0xFFFFu) << 16);          // tensor_dim0[15:0] @ bits 63:48
  g1[2] = (int)(((tile_d0 >> 16) & 0xFFFFu) |        // tensor_dim0[31:16]
                ((tile_d1 & 0xFFFFu) << 16));        // tensor_dim1[15:0]
  g1[3] = (int)(((tile_d1 >> 16) & 0xFFFFu) |        // tensor_dim1[31:16]
                ((tile_d0 & 0xFFFFu) << 16));        // tile_dim0
  g1[4] = (int)(tile_d1 & 0xFFFFu);                  // tile_dim1 (tile_dim2=0 -> 2D)
  g1[5] = (int)stride_d0_elems;                      // tensor_dim0_stride[31:0]
  g1[6] = 0;                                         // stride hi / dim1_stride lo
  g1[7] = 0;
  v4i gz4 = {0, 0, 0, 0};
  v8i gz8 = {0, 0, 0, 0, 0, 0, 0, 0};
  __builtin_amdgcn_tensor_load_to_lds(g0, g1, gz4, gz4, gz8, 0);
}
#else
#define HAVE_TDM 0
#endif

// ---------------------------------------------------------------------------
// Generic f16 GEMM: C[M,N] = A[M,K] * W[N,K]^T (+bias), f32 accumulate.
// ---------------------------------------------------------------------------
__global__ __launch_bounds__(256)
void gemm_f16(const _Float16* __restrict__ A, const _Float16* __restrict__ W,
              const float* __restrict__ bias, _Float16* __restrict__ C,
              int N, int K, int wrows) {
  int wave = threadIdx.x >> 5;
  int nn = lane_id() & 15;
  int wr = wave % wrows, wc = wave / wrows;
  int wcols = 8 / wrows;
  size_t row0 = (size_t)blockIdx.x * (wrows * 16) + wr * 16;
  int col0 = blockIdx.y * (wcols * 64) + wc * 64;

  v8f acc[4];
#pragma unroll
  for (int f = 0; f < 4; ++f) {
    float b0 = bias ? bias[col0 + f * 16 + nn] : 0.0f;
    v8f z = {};
#pragma unroll
    for (int i = 0; i < 8; ++i) z[i] = b0;
    acc[f] = z;
  }

  for (int k = 0; k < K; k += 32) {
    v16h a = load_a_frag(A + row0 * K + k, K);
#pragma unroll
    for (int f = 0; f < 4; ++f) {
      v16h b = load_b_frag(W + (size_t)(col0 + f * 16) * K + k, K);
      acc[f] = WMMA_F16(a, b, acc[f]);
    }
  }
#pragma unroll
  for (int f = 0; f < 4; ++f)
    store_c_f16(C + row0 * N + col0 + f * 16, N, acc[f]);
}

// ---------------------------------------------------------------------------
// Attention: one block per (batch b, head). N=128, DH=64, scale=1/8.
// ---------------------------------------------------------------------------
__global__ __launch_bounds__(256)
void attention_kernel(const _Float16* __restrict__ qkv,
                      const unsigned* __restrict__ maskbits, // [B*128, 4]
                      _Float16* __restrict__ ctx) {          // [B*128, 256]
  __shared__ _Float16 sk[128 * 64];
  __shared__ _Float16 sv[128 * 64];
  __shared__ _Float16 sattn[8 * 16 * 128];

  int b = blockIdx.x, head = blockIdx.y;
  const _Float16* base = qkv + (size_t)b * 128 * 768;

#if HAVE_TDM
  // Async DMA the K and V tiles (128 rows x 64 f16, row stride 768) into LDS.
  if (threadIdx.x < 32) {
    tdm_load_2d_f16((unsigned)(size_t)(void*)sk, base + 256 + head * 64, 64, 128, 768);
    tdm_load_2d_f16((unsigned)(size_t)(void*)sv, base + 512 + head * 64, 64, 128, 768);
    __builtin_amdgcn_s_wait_tensorcnt(0);
  }
#else
  for (int idx = threadIdx.x; idx < 1024; idx += blockDim.x) {
    int r = idx >> 3, o = (idx & 7) * 8;
    *(uint4*)(sk + r * 64 + o) = *(const uint4*)(base + (size_t)r * 768 + 256 + head * 64 + o);
    *(uint4*)(sv + r * 64 + o) = *(const uint4*)(base + (size_t)r * 768 + 512 + head * 64 + o);
  }
#endif
  __syncthreads();

  int wave = threadIdx.x >> 5;
  int nn = lane_id() & 15, hh = lane_id() >> 4;
  int q0 = wave * 16; // 16 query rows per wave

  // scores: 16 x 128 stripe, K-dim 64
  v8f acc[8];
#pragma unroll
  for (int f = 0; f < 8; ++f) { v8f z = {}; acc[f] = z; }
#pragma unroll
  for (int kf = 0; kf < 64; kf += 32) {
    v16h a = load_a_frag(base + (size_t)q0 * 768 + head * 64 + kf, 768);
#pragma unroll
    for (int f = 0; f < 8; ++f) {
      v16h bb = load_b_frag(sk + (size_t)(f * 16) * 64 + kf, 64);
      acc[f] = WMMA_F16(a, bb, acc[f]);
    }
  }

  // scale + mask (bit==1 means communicate, bit==0 -> -1e9)
  const float scale = 0.125f;
#pragma unroll
  for (int i = 0; i < 8; ++i) {
    int row = q0 + i + 8 * hh;
    uint4 w = ((const uint4*)maskbits)[(size_t)b * 128 + row];
    unsigned wv[4] = {w.x, w.y, w.z, w.w};
#pragma unroll
    for (int f = 0; f < 8; ++f) {
      int col = f * 16 + nn;
      unsigned bit = (wv[col >> 5] >> (col & 31)) & 1u;
      float s = acc[f][i] * scale;
      acc[f][i] = bit ? s : -1e9f;
    }
  }

  // softmax per row: row lives in one 16-lane half across the 8 frags
#pragma unroll
  for (int i = 0; i < 8; ++i) {
    float m = acc[0][i];
#pragma unroll
    for (int f = 1; f < 8; ++f) m = fmaxf(m, acc[f][i]);
#pragma unroll
    for (int x = 1; x < 16; x <<= 1) m = fmaxf(m, __shfl_xor(m, x, 32));
    float s = 0.0f;
#pragma unroll
    for (int f = 0; f < 8; ++f) { float e = __expf(acc[f][i] - m); acc[f][i] = e; s += e; }
#pragma unroll
    for (int x = 1; x < 16; x <<= 1) s += __shfl_xor(s, x, 32);
    float inv = 1.0f / s;
#pragma unroll
    for (int f = 0; f < 8; ++f) acc[f][i] *= inv;
  }

  // write probs to LDS (wave-private region) to re-shape into A fragments
  _Float16* sa = sattn + (size_t)wave * (16 * 128);
#pragma unroll
  for (int i = 0; i < 8; ++i)
#pragma unroll
    for (int f = 0; f < 8; ++f)
      sa[(size_t)(i + 8 * hh) * 128 + f * 16 + nn] = (_Float16)acc[f][i];
  __syncthreads();

  // ctx = attn(16x128) x V(128x64)
  v8f o[4];
#pragma unroll
  for (int f = 0; f < 4; ++f) { v8f z = {}; o[f] = z; }
#pragma unroll
  for (int kk = 0; kk < 128; kk += 32) {
    v16h a = load_a_frag(sa + kk, 128);
#pragma unroll
    for (int f = 0; f < 4; ++f) {
      v16h bb = load_b_frag_kn(sv + (size_t)kk * 64 + f * 16, 64);
      o[f] = WMMA_F16(a, bb, o[f]);
    }
  }
#pragma unroll
  for (int f = 0; f < 4; ++f)
    store_c_f16(ctx + ((size_t)b * 128 + q0) * 256 + head * 64 + f * 16, 256, o[f]);
}

// ---------------------------------------------------------------------------
// Fused GRU: per block 16 rows; gi = info@Wih^T+bih, gh = lat@Whh^T+bhh (WMMA),
// gates via LDS exchange, masked update of latent (f32 + optional f16).
// ---------------------------------------------------------------------------
__global__ __launch_bounds__(256)
void gru_fused(const _Float16* __restrict__ info16, // [M,64]
               _Float16* __restrict__ lat16,        // [M,256] read + in-place f16 update
               const float* __restrict__ latin,     // [M,256] f32 old latent
               const _Float16* __restrict__ Wih16,  // [768,64]
               const _Float16* __restrict__ Whh16,  // [768,256]
               const float* __restrict__ bih, const float* __restrict__ bhh,
               const unsigned char* __restrict__ upmask, // [M]
               float* __restrict__ latout,          // [M,256]
               int write_f16) {
  __shared__ float sgi[16 * 768];
  __shared__ float sgh[16 * 768];

  size_t row0 = (size_t)blockIdx.x * 16;
  int wave = threadIdx.x >> 5;
  int nn = lane_id() & 15, hh = lane_id() >> 4;
  int col0 = wave * 96; // 6 col tiles per wave, 8 waves -> 768

  v8f gi[6], gh[6];
#pragma unroll
  for (int f = 0; f < 6; ++f) { v8f z = {}; gi[f] = z; gh[f] = z; }

#pragma unroll
  for (int k = 0; k < 64; k += 32) {
    v16h a = load_a_frag(info16 + row0 * 64 + k, 64);
#pragma unroll
    for (int f = 0; f < 6; ++f) {
      v16h b = load_b_frag(Wih16 + (size_t)(col0 + f * 16) * 64 + k, 64);
      gi[f] = WMMA_F16(a, b, gi[f]);
    }
  }
  for (int k = 0; k < 256; k += 32) {
    v16h a = load_a_frag(lat16 + row0 * 256 + k, 256);
#pragma unroll
    for (int f = 0; f < 6; ++f) {
      v16h b = load_b_frag(Whh16 + (size_t)(col0 + f * 16) * 256 + k, 256);
      gh[f] = WMMA_F16(a, b, gh[f]);
    }
  }

#pragma unroll
  for (int f = 0; f < 6; ++f) {
    int col = col0 + f * 16 + nn;
    float bi = bih[col], bh = bhh[col];
#pragma unroll
    for (int i = 0; i < 8; ++i) {
      sgi[(size_t)(i + 8 * hh) * 768 + col] = gi[f][i] + bi;
      sgh[(size_t)(i + 8 * hh) * 768 + col] = gh[f][i] + bh;
    }
  }
  __syncthreads();

  for (int t = threadIdx.x; t < 16 * 256; t += blockDim.x) {
    int r = t >> 8, d = t & 255;
    float ir  = sgi[(size_t)r * 768 + d],       hr = sgh[(size_t)r * 768 + d];
    float iz  = sgi[(size_t)r * 768 + 256 + d], hz = sgh[(size_t)r * 768 + 256 + d];
    float inn = sgi[(size_t)r * 768 + 512 + d], hn = sgh[(size_t)r * 768 + 512 + d];
    float rg = 1.0f / (1.0f + __expf(-(ir + hr)));
    float zg = 1.0f / (1.0f + __expf(-(iz + hz)));
    float ng = tanhf(inn + rg * hn);
    size_t g = (row0 + r) * 256 + d;
    float hold = latin[g];
    float v = upmask[row0 + r] ? ((1.0f - zg) * ng + zg * hold) : hold;
    latout[g] = v;
    if (write_f16) lat16[g] = (_Float16)v;
  }
}

// ---------------------------------------------------------------------------
// Helpers: conversions + mask packing
// ---------------------------------------------------------------------------
__global__ void f32_to_f16_k(const float* __restrict__ src, _Float16* __restrict__ dst, size_t n) {
  for (size_t i = blockIdx.x * (size_t)blockDim.x + threadIdx.x; i < n;
       i += (size_t)gridDim.x * blockDim.x)
    dst[i] = (_Float16)src[i];
}
__global__ void copy_f32_k(const float* __restrict__ src, float* __restrict__ dst, size_t n) {
  for (size_t i = blockIdx.x * (size_t)blockDim.x + threadIdx.x; i < n;
       i += (size_t)gridDim.x * blockDim.x)
    dst[i] = src[i];
}

// one block (128 threads) per (b,n) row: pack 128 mask ints -> 4 words + update flag
__global__ __launch_bounds__(128)
void pack_mask(const int* __restrict__ comm, unsigned* __restrict__ bits,
               unsigned char* __restrict__ upmask) {
  __shared__ unsigned sw[4];
  size_t row = blockIdx.x;
  if (threadIdx.x < 4) sw[threadIdx.x] = 0u;
  __syncthreads();
  int lane = threadIdx.x & 31, wave = threadIdx.x >> 5;
  int m = comm[row * 128 + threadIdx.x];
  if (m != 0) atomicOr(&sw[wave], 1u << lane);
  __syncthreads();
  if (threadIdx.x == 0) {
    unsigned a = sw[0], b = sw[1], c = sw[2], d = sw[3];
    uint4 v; v.x = a; v.y = b; v.z = c; v.w = d;
    ((uint4*)bits)[row] = v;
    upmask[row] = (unsigned char)((__popc(a) + __popc(b) + __popc(c) + __popc(d)) > 1 ? 1 : 0);
  }
}

// ---------------------------------------------------------------------------
// Workspace layout (bytes)
// ---------------------------------------------------------------------------
static constexpr size_t BN      = 131072;            // B*N tokens
static constexpr size_t OFF_LAT16  = 0;                                // 64 MB
static constexpr size_t OFF_QKV16  = OFF_LAT16  + BN * 256 * 2;        // 192 MB
static constexpr size_t OFF_CTX16  = OFF_QKV16  + BN * 768 * 2;        // 64 MB
static constexpr size_t OFF_INFO16 = OFF_CTX16  + BN * 256 * 2;        // 16 MB
static constexpr size_t OFF_LAT32A = OFF_INFO16 + BN * 64 * 2;         // 128 MB
static constexpr size_t OFF_WQKV   = OFF_LAT32A + BN * 256 * 4;
static constexpr size_t OFF_WO     = OFF_WQKV   + 768 * 256 * 2;
static constexpr size_t OFF_WIH    = OFF_WO     + 64 * 256 * 2;
static constexpr size_t OFF_WHH    = OFF_WIH    + 768 * 64 * 2;
static constexpr size_t OFF_BQKV   = OFF_WHH    + 768 * 256 * 2;
static constexpr size_t OFF_BITS   = OFF_BQKV   + 768 * 4;
static constexpr size_t OFF_UPM    = OFF_BITS   + BN * 4 * 4;

extern "C" void kernel_launch(void* const* d_in, const int* in_sizes, int n_in,
                              void* d_out, int out_size, void* d_ws, size_t ws_size,
                              hipStream_t stream) {
  (void)in_sizes; (void)n_in; (void)out_size; (void)ws_size;
  const float* latent = (const float*)d_in[0];
  const int*   comm   = (const int*)d_in[1];
  const float* Wq  = (const float*)d_in[3];
  const float* bq  = (const float*)d_in[4];
  const float* Wk  = (const float*)d_in[5];
  const float* bk  = (const float*)d_in[6];
  const float* Wv  = (const float*)d_in[7];
  const float* bv  = (const float*)d_in[8];
  const float* Wo  = (const float*)d_in[9];
  const float* Wih = (const float*)d_in[10];
  const float* Whh = (const float*)d_in[11];
  const float* bih = (const float*)d_in[12];
  const float* bhh = (const float*)d_in[13];

  char* ws = (char*)d_ws;
  _Float16* lat16  = (_Float16*)(ws + OFF_LAT16);
  _Float16* qkv16  = (_Float16*)(ws + OFF_QKV16);
  _Float16* ctx16  = (_Float16*)(ws + OFF_CTX16);
  _Float16* info16 = (_Float16*)(ws + OFF_INFO16);
  float*    lat32a = (float*)(ws + OFF_LAT32A);
  _Float16* Wqkv16 = (_Float16*)(ws + OFF_WQKV);
  _Float16* Wo16   = (_Float16*)(ws + OFF_WO);
  _Float16* Wih16  = (_Float16*)(ws + OFF_WIH);
  _Float16* Whh16  = (_Float16*)(ws + OFF_WHH);
  float*    bqkv   = (float*)(ws + OFF_BQKV);
  unsigned* bits   = (unsigned*)(ws + OFF_BITS);
  unsigned char* upmask = (unsigned char*)(ws + OFF_UPM);

  // ---- precompute (every call; deterministic) ----
  f32_to_f16_k<<<2048, 256, 0, stream>>>(latent, lat16, BN * 256);
  f32_to_f16_k<<<64, 256, 0, stream>>>(Wq, Wqkv16,               256 * 256);
  f32_to_f16_k<<<64, 256, 0, stream>>>(Wk, Wqkv16 + 256 * 256,   256 * 256);
  f32_to_f16_k<<<64, 256, 0, stream>>>(Wv, Wqkv16 + 512 * 256,   256 * 256);
  f32_to_f16_k<<<16, 256, 0, stream>>>(Wo, Wo16,   64 * 256);
  f32_to_f16_k<<<48, 256, 0, stream>>>(Wih, Wih16, 768 * 64);
  f32_to_f16_k<<<192, 256, 0, stream>>>(Whh, Whh16, 768 * 256);
  copy_f32_k<<<1, 256, 0, stream>>>(bq, bqkv,       256);
  copy_f32_k<<<1, 256, 0, stream>>>(bk, bqkv + 256, 256);
  copy_f32_k<<<1, 256, 0, stream>>>(bv, bqkv + 512, 256);
  pack_mask<<<131072, 128, 0, stream>>>(comm, bits, upmask);

  // ---- 2 comm layers ----
  for (int layer = 0; layer < 2; ++layer) {
    const float* lin  = (layer == 0) ? latent : lat32a;
    float*       lout = (layer == 0) ? lat32a : (float*)d_out;
    // QKV: [BN,256] x [768,256]^T -> [BN,768]
    gemm_f16<<<dim3(2048, 6), 256, 0, stream>>>(lat16, Wqkv16, bqkv, qkv16, 768, 256, 4);
    // attention per (batch, head)
    attention_kernel<<<dim3(1024, 4), 256, 0, stream>>>(qkv16, bits, ctx16);
    // output proj: [BN,256] x [64,256]^T -> [BN,64]
    gemm_f16<<<dim3(1024, 1), 256, 0, stream>>>(ctx16, Wo16, nullptr, info16, 64, 256, 8);
    // fused GRU + masked update
    gru_fused<<<8192, 256, 0, stream>>>(info16, lat16, lin, Wih16, Whh16, bih, bhh,
                                        upmask, lout, (layer == 0) ? 1 : 0);
  }
}
